// SRNN_14345190768815
// MI455X (gfx1250) — matrix operand order repaired and checked by reading, not compiled, not run on hardware
//
#include <hip/hip_runtime.h>

// ---------------------------------------------------------------------------
// SRNN for MI455X (gfx1250, wave32, WMMA 16x16x32 f16).
//
// Phase 1 (srnn_prep):  repack weights fp32 -> f16 in WMMA-B-fragment layout,
//                       zero-padded (100->112/128, 50->64); pad biases.
// Phase 2 (srnn_gate):  262144 rows x {128->256 and 128->100->100->100->50->256}
//                       via v_wmma_f32_16x16x32_f16. Two 16-row tiles per wave
//                       share each B fragment; x fragments built directly from
//                       global (nontemporal) loads; gate stored f16.
// Phase 3 (srnn_scan):  1024-step rolled-ReLU scan, LDS double buffer,
//                       fused sigmoid(hidden @ Wo + bo) head.
// ---------------------------------------------------------------------------

#define Bb 256
#define Tt 1024
#define Dd 128
#define Hh 256

typedef __attribute__((ext_vector_type(16))) _Float16 v16h;
typedef __attribute__((ext_vector_type(8)))  _Float16 v8h;
typedef __attribute__((ext_vector_type(8)))  float    v8f;
typedef __attribute__((ext_vector_type(4)))  float    v4f;

constexpr size_t GATE_HALFS = (size_t)Tt * Bb * Hh;       // 67,108,864
constexpr size_t GATE_BYTES = GATE_HALFS * 2;             // 134,217,728
// half-offsets inside the weight block (each frag = 512 halfs = 1KB):
constexpr int WX_OFF = 0;        // 4 Kchunks x 16 Ntiles  -> 32768 halfs
constexpr int W1_OFF = 32768;    // 4 x 7                  -> 14336
constexpr int W2_OFF = 47104;    // 4 x 7
constexpr int W3_OFF = 61440;    // 4 x 7
constexpr int W4_OFF = 75776;    // 4 x 4                  -> 8192
constexpr int W5_OFF = 83968;    // 2 x 16                 -> 16384
constexpr int WHALFS = 100352;
constexpr int BIAS_FLOATS = 912;                          // 256+112+112+112+64+256
constexpr int WBLK_BYTES = WHALFS * 2 + BIAS_FLOATS * 4;  // 204,352 (16B multiple)
constexpr int SCR_HALFS = 16 * Dd;                        // per-tile LDS scratch

// ---------------------------------------------------------------------------
// Phase 1: weight repack into WMMA B-fragment layout.
// B fragment (32Kx16N f16, 512 halfs): lane l<16 holds N=l, K=kb+{0..7,16..23};
// lane l>=16 holds N=l-16, K=kb+{8..15,24..31} (32B per lane).
// ---------------------------------------------------------------------------
__device__ inline void pack_frag(_Float16* dst, const float* src,
                                 int srcK, int srcN, int NT, int li) {
  int f    = li >> 9;
  int r    = li & 511;
  int lane = r >> 4;
  int e    = r & 15;
  int kc   = f / NT;
  int j    = f - kc * NT;
  int k    = kc * 32 + ((lane & 16) ? 8 : 0) + ((e < 8) ? e : e + 8);
  int n    = j * 16 + (lane & 15);
  float v  = (k < srcK && n < srcN) ? src[k * srcN + n] : 0.0f;
  dst[li]  = (_Float16)v;
}

__global__ void srnn_prep(const float* Wx, const float* bx,
                          const float* W1, const float* b1,
                          const float* W2, const float* b2,
                          const float* W3, const float* b3,
                          const float* W4, const float* b4,
                          const float* W5, const float* b5,
                          _Float16* wblk) {
  int idx = blockIdx.x * blockDim.x + threadIdx.x;
  if (idx < 32768)        pack_frag(wblk + WX_OFF, Wx, 128, 256, 16, idx);
  else if (idx < 47104)   pack_frag(wblk + W1_OFF, W1, 128, 100, 7,  idx - 32768);
  else if (idx < 61440)   pack_frag(wblk + W2_OFF, W2, 100, 100, 7,  idx - 47104);
  else if (idx < 75776)   pack_frag(wblk + W3_OFF, W3, 100, 100, 7,  idx - 61440);
  else if (idx < 83968)   pack_frag(wblk + W4_OFF, W4, 100, 50,  4,  idx - 75776);
  else if (idx < 100352)  pack_frag(wblk + W5_OFF, W5, 50,  256, 16, idx - 83968);
  else if (idx < 100352 + BIAS_FLOATS) {
    float* bp = (float*)(wblk + WHALFS);
    int bi = idx - 100352;
    float v;
    if (bi < 256)       { v = bx[bi]; }
    else if (bi < 368)  { int c = bi - 256; v = (c < 100) ? b1[c] : 0.0f; }
    else if (bi < 480)  { int c = bi - 368; v = (c < 100) ? b2[c] : 0.0f; }
    else if (bi < 592)  { int c = bi - 480; v = (c < 100) ? b3[c] : 0.0f; }
    else if (bi < 656)  { int c = bi - 592; v = (c < 50)  ? b4[c] : 0.0f; }
    else                { v = b5[bi - 656]; }
    bp[bi] = v;
  }
}

// ---------------------------------------------------------------------------
// Phase 2 helpers.
// ISA 16-bit A layout: lane l<16 -> row l, K = kc*32 + {0..7, 16..23};
// lane l>=16 -> row l-16, K = kc*32 + {8..15, 24..31}.
// ---------------------------------------------------------------------------
__device__ inline v16h cvt16(v4f a0, v4f a1, v4f c0, v4f c1) {
  v16h r;
  r[0]  = (_Float16)a0[0]; r[1]  = (_Float16)a0[1];
  r[2]  = (_Float16)a0[2]; r[3]  = (_Float16)a0[3];
  r[4]  = (_Float16)a1[0]; r[5]  = (_Float16)a1[1];
  r[6]  = (_Float16)a1[2]; r[7]  = (_Float16)a1[3];
  r[8]  = (_Float16)c0[0]; r[9]  = (_Float16)c0[1];
  r[10] = (_Float16)c0[2]; r[11] = (_Float16)c0[3];
  r[12] = (_Float16)c1[0]; r[13] = (_Float16)c1[1];
  r[14] = (_Float16)c1[2]; r[15] = (_Float16)c1[3];
  return r;
}

// A fragment straight from global fp32 x (two 32B segments per lane), NT loads.
__device__ inline v16h loadA_gx(const float* p) {
  v4f a0 = __builtin_nontemporal_load((const v4f*)(p));
  v4f a1 = __builtin_nontemporal_load((const v4f*)(p + 4));
  v4f c0 = __builtin_nontemporal_load((const v4f*)(p + 16));
  v4f c1 = __builtin_nontemporal_load((const v4f*)(p + 20));
  return cvt16(a0, a1, c0, c1);
}

// A fragment from per-wave LDS scratch [16 rows][128 K] (2x ds_load_b128).
__device__ inline v16h loadA_lds(const _Float16* scr, int kc, int lane) {
  int row = lane & 15;
  int kb  = kc * 32 + ((lane & 16) ? 8 : 0);
  const _Float16* p = scr + row * Dd + kb;
  v8h lo = *(const v8h*)(p);
  v8h hi = *(const v8h*)(p + 16);
  return __builtin_shufflevector(lo, hi, 0, 1, 2, 3, 4, 5, 6, 7,
                                 8, 9, 10, 11, 12, 13, 14, 15);
}

__device__ inline v8f wmma16(v16h a, v16h b, v8f c) {
  return __builtin_amdgcn_wmma_f32_16x16x32_f16(false, a, false, b, (short)0,
                                                c, false, false);
}

// One MLP layer for TWO 16-row tiles sharing each B fragment.
template <int KC, int NT>
__device__ inline void mlp_stage2(const v16h* a0, const v16h* a1,
                                  const _Float16* Wf, const float* bp,
                                  _Float16* s0, _Float16* s1, int lane) {
  int nloc = lane & 15;
  int mo   = (lane & 16) ? 8 : 0;
#pragma unroll
  for (int j = 0; j < NT; ++j) {
    v16h bf[KC];
#pragma unroll
    for (int k = 0; k < KC; ++k)
      bf[k] = *(const v16h*)(Wf + ((k * NT + j) << 9) + (lane << 4));
    v8f acc0 = {}, acc1 = {};
#pragma unroll
    for (int k = 0; k < KC; ++k) {
      acc0 = wmma16(a0[k], bf[k], acc0);
      acc1 = wmma16(a1[k], bf[k], acc1);
    }
    float bias = bp[j * 16 + nloc];
#pragma unroll
    for (int e = 0; e < 8; ++e) {
      float v0 = acc0[e] + bias;
      float v1 = acc1[e] + bias;
      v0 = (v0 > 0.0f) ? v0 : 0.0f;
      v1 = (v1 > 0.0f) ? v1 : 0.0f;
      s0[(e + mo) * Dd + j * 16 + nloc] = (_Float16)v0;
      s1[(e + mo) * Dd + j * 16 + nloc] = (_Float16)v1;
    }
  }
}

__global__ void __launch_bounds__(256)
srnn_gate(const float* __restrict__ x, const _Float16* __restrict__ wblk,
          _Float16* __restrict__ gate) {
  extern __shared__ char smem[];
  _Float16* wl = (_Float16*)smem;
  const float* bl = (const float*)(wl + WHALFS);
  _Float16* scrAll = (_Float16*)(smem + WBLK_BYTES);

  int tid  = threadIdx.x;
  int wave = tid >> 5;
  int lane = tid & 31;

  // Stage all packed weights + biases into LDS once per workgroup (204,352 B).
  {
    const uint4* s = (const uint4*)wblk;
    uint4* d = (uint4*)smem;
    for (int i = tid; i < WBLK_BYTES / 16; i += 256) d[i] = s[i];
  }
  __syncthreads();

  _Float16* scr0 = scrAll + wave * (2 * SCR_HALFS);
  _Float16* scr1 = scr0 + SCR_HALFS;
  const float* bxl = bl;
  const float* b1l = bl + 256;
  const float* b2l = bl + 368;
  const float* b3l = bl + 480;
  const float* b4l = bl + 592;
  const float* b5l = bl + 656;

  int nloc = lane & 15;
  int mo   = (lane & 16) ? 8 : 0;
  int koff = (lane & 16) ? 8 : 0;

  int pairBase = blockIdx.x * 16;       // 512 WGs x 16 pairs = 8192 pairs
#pragma unroll 1
  for (int it = 0; it < 2; ++it) {
    int pair = pairBase + wave + it * 8;
    int n0 = pair << 5;                 // 32 rows (2 tiles) share t
    int t  = n0 >> 8;
    int b0 = n0 & 255;

    // x A-fragments straight from global (32x global_load_b128, NT hint).
    const float* xb0 = x + (size_t)(b0 + nloc) * (Tt * Dd) + (size_t)t * Dd + koff;
    const float* xb1 = xb0 + (size_t)16 * (Tt * Dd);
    v16h ax0[4], ax1[4];
#pragma unroll
    for (int k = 0; k < 4; ++k) {
      ax0[k] = loadA_gx(xb0 + k * 32);
      ax1[k] = loadA_gx(xb1 + k * 32);
    }

    mlp_stage2<4, 7>(ax0, ax1, wl + W1_OFF, b1l, scr0, scr1, lane);  // 128->112
    v16h a0[4], a1[4];
#pragma unroll
    for (int k = 0; k < 4; ++k) { a0[k] = loadA_lds(scr0, k, lane);
                                  a1[k] = loadA_lds(scr1, k, lane); }
    mlp_stage2<4, 7>(a0, a1, wl + W2_OFF, b2l, scr0, scr1, lane);    // 112->112
#pragma unroll
    for (int k = 0; k < 4; ++k) { a0[k] = loadA_lds(scr0, k, lane);
                                  a1[k] = loadA_lds(scr1, k, lane); }
    mlp_stage2<4, 7>(a0, a1, wl + W3_OFF, b3l, scr0, scr1, lane);    // 112->112
#pragma unroll
    for (int k = 0; k < 4; ++k) { a0[k] = loadA_lds(scr0, k, lane);
                                  a1[k] = loadA_lds(scr1, k, lane); }
    mlp_stage2<4, 4>(a0, a1, wl + W4_OFF, b4l, scr0, scr1, lane);    // 112->64

    v16h a50[2], a51[2];
#pragma unroll
    for (int k = 0; k < 2; ++k) { a50[k] = loadA_lds(scr0, k, lane);
                                  a51[k] = loadA_lds(scr1, k, lane); }

    // Final: gate = sigmoid(h4@W5 + b5) * sigmoid(x@Wx + bx), f16 out.
#pragma unroll
    for (int j = 0; j < 16; ++j) {
      v16h bf5[2], bfx[4];
#pragma unroll
      for (int k = 0; k < 2; ++k)
        bf5[k] = *(const v16h*)(wl + W5_OFF + ((k * 16 + j) << 9) + (lane << 4));
#pragma unroll
      for (int k = 0; k < 4; ++k)
        bfx[k] = *(const v16h*)(wl + WX_OFF + ((k * 16 + j) << 9) + (lane << 4));
      v8f aF0 = {}, aF1 = {}, aL0 = {}, aL1 = {};
#pragma unroll
      for (int k = 0; k < 2; ++k) {
        aF0 = wmma16(a50[k], bf5[k], aF0);
        aF1 = wmma16(a51[k], bf5[k], aF1);
      }
#pragma unroll
      for (int k = 0; k < 4; ++k) {
        aL0 = wmma16(ax0[k], bfx[k], aL0);
        aL1 = wmma16(ax1[k], bfx[k], aL1);
      }
      int h = (j << 4) + nloc;
      float b5v = b5l[h];
      float bxv = bxl[h];
      size_t base = (size_t)t * (Bb * Hh) + (size_t)(b0 + mo) * Hh + h;
#pragma unroll
      for (int e = 0; e < 8; ++e) {
        float fx0 = 1.0f / (1.0f + __expf(-(aF0[e] + b5v)));
        float lx0 = 1.0f / (1.0f + __expf(-(aL0[e] + bxv)));
        float fx1 = 1.0f / (1.0f + __expf(-(aF1[e] + b5v)));
        float lx1 = 1.0f / (1.0f + __expf(-(aL1[e] + bxv)));
        gate[base + (size_t)e * Hh]        = (_Float16)(fx0 * lx0);
        gate[base + (size_t)(e + 16) * Hh] = (_Float16)(fx1 * lx1);
      }
    }
  }
}

// ---------------------------------------------------------------------------
// Phase 3: sequential rolled-ReLU scan + output head.
// ---------------------------------------------------------------------------
__global__ void __launch_bounds__(256)
srnn_scan(const _Float16* __restrict__ gate, const float* __restrict__ Wo,
          const float* __restrict__ bo, const int* __restrict__ shiftp,
          float* __restrict__ out) {
  __shared__ float buf[2][Hh];
  __shared__ float red[Hh];
  int b = blockIdx.x;
  int h = threadIdx.x;
  int s = ((*shiftp) % Hh + Hh) % Hh;
  int src = (h - s + Hh) % Hh;          // roll(hid, s)[h] = hid[h - s]

  buf[1][h] = 0.0f;                     // state "after step -1"
  __syncthreads();

  const _Float16* gp = gate + (size_t)b * Hh + h;
#pragma unroll 1
  for (int t = 0; t < Tt; ++t) {
    float g = (float)gp[(size_t)t * (Bb * Hh)];
    if (t + 8 < Tt)
      __builtin_prefetch(gp + (size_t)(t + 8) * (Bb * Hh), 0, 0);
    float prev = buf[(t + 1) & 1][src];
    float v = prev + g;
    v = (v > 0.0f) ? v : 0.0f;
    buf[t & 1][h] = v;
    __syncthreads();
  }

  float hv = buf[(Tt - 1) & 1][h];
  out[Bb + b * Hh + h] = hv;            // hidden, after output[B]

  red[h] = hv * Wo[h];
  __syncthreads();
#pragma unroll
  for (int off = 128; off > 0; off >>= 1) {
    if (h < off) red[h] += red[h + off];
    __syncthreads();
  }
  if (h == 0) out[b] = 1.0f / (1.0f + __expf(-(red[0] + bo[0])));
}

// ---------------------------------------------------------------------------
extern "C" void kernel_launch(void* const* d_in, const int* in_sizes, int n_in,
                              void* d_out, int out_size, void* d_ws,
                              size_t ws_size, hipStream_t stream) {
  const float* x  = (const float*)d_in[0];
  const float* Wx = (const float*)d_in[1];
  const float* bx = (const float*)d_in[2];
  const float* W1 = (const float*)d_in[3];
  const float* b1 = (const float*)d_in[4];
  const float* W2 = (const float*)d_in[5];
  const float* b2 = (const float*)d_in[6];
  const float* W3 = (const float*)d_in[7];
  const float* b3 = (const float*)d_in[8];
  const float* W4 = (const float*)d_in[9];
  const float* b4 = (const float*)d_in[10];
  const float* W5 = (const float*)d_in[11];
  const float* b5 = (const float*)d_in[12];
  const float* Wo = (const float*)d_in[13];
  const float* bo = (const float*)d_in[14];
  const int* shiftp = (const int*)d_in[15];

  _Float16* gate = (_Float16*)d_ws;
  _Float16* wblk = (_Float16*)((char*)d_ws + GATE_BYTES);

  srnn_prep<<<(100352 + BIAS_FLOATS + 255) / 256, 256, 0, stream>>>(
      Wx, bx, W1, b1, W2, b2, W3, b3, W4, b4, W5, b5, wblk);

  size_t smem = (size_t)WBLK_BYTES + 8 * 2 * SCR_HALFS * sizeof(_Float16); // 269,888 B
  srnn_gate<<<512, 256, smem, stream>>>(x, wblk, gate);

  srnn_scan<<<256, 256, 0, stream>>>(gate, Wo, bo, shiftp, (float*)d_out);
}
